// LEFTNet_72868415144419
// MI455X (gfx1250) — compile-verified, matrix operand order checked by reading, not run on hardware
//
#include <hip/hip_runtime.h>
#include <math.h>

#define Hh 128
#define H3 384
#define KW 416   // 3H + R
#define Rr 32

typedef __attribute__((ext_vector_type(16))) __bf16        v16bf;
typedef __attribute__((ext_vector_type(8)))  float         v8f;
typedef __attribute__((ext_vector_type(8)))  unsigned int  v8u;

// ---------- helpers ----------

__device__ __forceinline__ unsigned int f2bf(float f) {
  unsigned int u = __float_as_uint(f);
  if ((u & 0x7F800000u) == 0x7F800000u) return (u >> 16) & 0xFFFFu; // Inf/NaN passthrough
  unsigned int r = (u + 0x7FFFu + ((u >> 16) & 1u)) >> 16;          // RNE
  return r & 0xFFFFu;
}

__device__ __forceinline__ v8f v8f_zero() {
  v8f z = {0.f, 0.f, 0.f, 0.f, 0.f, 0.f, 0.f, 0.f};
  return z;
}

__device__ __forceinline__ float silu(float v) { return v / (1.f + __expf(-v)); }

// A fragment (16x32 bf16, MxK) from LDS row-major bf16 tile (stride in elements, even)
__device__ __forceinline__ v16bf load_a_lds(const unsigned short* s, int row0, int stride,
                                            int koff, int lane) {
  int row = row0 + (lane & 15);
  int hf  = lane >> 4;
  const unsigned int* p = (const unsigned int*)(s + (size_t)row * stride + koff);
  v8u u;
#pragma unroll
  for (int v = 0; v < 8; ++v) {
    int d = ((v >> 2) << 3) + (hf << 2) + (v & 3); // dword index within 32-wide K tile
    u[v] = p[d];
  }
  return __builtin_bit_cast(v16bf, u);
}

// B fragment (32x16 bf16, KxN) from pre-swizzled global weights:
// dword index = ((nt*KT + kt)*32 + lane)*8 + v  -> per-lane 32B contiguous
__device__ __forceinline__ v16bf load_b_swz(const unsigned int* bs, int nt, int kt, int KT,
                                            int lane) {
  size_t idx = ((size_t)(nt * KT + kt) * 32u + (unsigned)lane) * 8u;
  v8u u = *(const v8u*)(bs + idx);
  return __builtin_bit_cast(v16bf, u);
}

__device__ __forceinline__ v8f wmma_bf16(v16bf a, v16bf b, v8f c) {
  return __builtin_amdgcn_wmma_f32_16x16x32_bf16(false, a, false, b, (short)0, c, false, false);
}

__device__ __forceinline__ void atomAddF(float* p, float v) {
  unsafeAtomicAdd(p, v);  // hardware global_atomic_add_f32
}

// ---------- kernels ----------

__global__ void zero_kernel(float* __restrict__ p, size_t n) {
  size_t i  = (size_t)blockIdx.x * blockDim.x + threadIdx.x;
  size_t st = (size_t)gridDim.x * blockDim.x;
  for (; i < n; i += st) p[i] = 0.f;
}

// Convert fp32 weight [K x N] (row-major, K = contraction dim) into WMMA-B-fragment order bf16
__global__ void swz_b_kernel(const float* __restrict__ W, unsigned int* __restrict__ out,
                             int K, int N) {
  int KT = K >> 5;
  int total = (N >> 4) * KT * 256;  // dwords
  for (int t = blockIdx.x * blockDim.x + threadIdx.x; t < total; t += gridDim.x * blockDim.x) {
    int v    = t & 7;
    int lane = (t >> 3) & 31;
    int tile = t >> 8;
    int kt = tile % KT;
    int nt = tile / KT;
    int hf   = lane >> 4;
    int colN = (nt << 4) + (lane & 15);
    int k0 = (kt << 5) + ((v >> 2) << 4) + (hf << 3) + ((v & 3) << 1);
    unsigned int lo = f2bf(W[(size_t)k0 * N + colN]);
    unsigned int hi = f2bf(W[(size_t)(k0 + 1) * N + colN]);
    out[t] = lo | (hi << 16);
  }
}

// xh = silu(x @ xw1 + xb1) @ xw2 + xb2    [N,384] fp32
__global__ __launch_bounds__(256) void node_proj_kernel(
    const float* __restrict__ x,
    const unsigned int* __restrict__ xw1s, const float* __restrict__ xb1,
    const unsigned int* __restrict__ xw2s, const float* __restrict__ xb2,
    float* __restrict__ xh, int nNodes) {
  __shared__ unsigned short sT[64 * 130];  // x tile / then t tile, bf16, stride 130
  const int tid = threadIdx.x, lane = tid & 31, wv = tid >> 5;
  const int row0 = blockIdx.x * 64;

  for (int i = tid; i < 64 * Hh; i += 256) {
    int r = i >> 7, c = i & (Hh - 1);
    int gr = row0 + r;
    float v = (gr < nNodes) ? x[(size_t)gr * Hh + c] : 0.f;
    sT[r * 130 + c] = (unsigned short)f2bf(v);
  }
  __syncthreads();

  // GEMM1: 64x128, wave wv owns column tile wv
  v8f acc[4];
#pragma unroll
  for (int rt = 0; rt < 4; ++rt) acc[rt] = v8f_zero();
#pragma unroll
  for (int kt = 0; kt < 4; ++kt) {
    v16bf b = load_b_swz(xw1s, wv, kt, 4, lane);
#pragma unroll
    for (int rt = 0; rt < 4; ++rt) {
      v16bf a = load_a_lds(sT, rt * 16, 130, kt * 32, lane);
      acc[rt] = wmma_bf16(a, b, acc[rt]);
    }
  }
  int col1 = (wv << 4) + (lane & 15);
  float b1v = xb1[col1];
  __syncthreads();
#pragma unroll
  for (int rt = 0; rt < 4; ++rt) {
#pragma unroll
    for (int i = 0; i < 8; ++i) {
      int row = rt * 16 + ((lane >> 4) << 3) + i;
      sT[row * 130 + col1] = (unsigned short)f2bf(silu(acc[rt][i] + b1v));
    }
  }
  __syncthreads();

  // GEMM2: 64x384
#pragma unroll
  for (int j = 0; j < 3; ++j) {
    int ct = wv + 8 * j;
    v8f a2[4];
#pragma unroll
    for (int rt = 0; rt < 4; ++rt) a2[rt] = v8f_zero();
#pragma unroll
    for (int kt = 0; kt < 4; ++kt) {
      v16bf b = load_b_swz(xw2s, ct, kt, 4, lane);
#pragma unroll
      for (int rt = 0; rt < 4; ++rt) {
        v16bf a = load_a_lds(sT, rt * 16, 130, kt * 32, lane);
        a2[rt] = wmma_bf16(a, b, a2[rt]);
      }
    }
    int col = (ct << 4) + (lane & 15);
    float b2v = xb2[col];
#pragma unroll
    for (int rt = 0; rt < 4; ++rt) {
#pragma unroll
      for (int i = 0; i < 8; ++i) {
        int row = rt * 16 + ((lane >> 4) << 3) + i;
        int gr = row0 + row;
        if (gr < nNodes) xh[(size_t)gr * H3 + col] = a2[rt][i] + b2v;
      }
    }
  }
}

// Per-edge: w = silu(weight@iw1+ib1)@iw2+ib2 ; rbfh = (rbf@rw+rb)*w ; m = xh[src]*rbfh ;
// scatter dx += m[:,0:128] ; dvec += (vec[src]*m1/sqrt3 + m2*edge_vector)/sqrtH
__global__ __launch_bounds__(256) void edge_msg_kernel(
    const float* __restrict__ wgt, const float* __restrict__ erbf,
    const int* __restrict__ srcI, const int* __restrict__ dstI,
    const float* __restrict__ evec,
    const unsigned int* __restrict__ iw1s, const float* __restrict__ ib1,
    const unsigned int* __restrict__ iw2s, const float* __restrict__ ib2,
    const unsigned int* __restrict__ rws, const float* __restrict__ rb,
    const float* __restrict__ xh, const float* __restrict__ vec,
    float* __restrict__ dx, float* __restrict__ dvec, int nE) {
  __shared__ unsigned short sA[64 * 418];  // weight tile bf16 (stride 418), reused for h1 (stride 386)
  __shared__ unsigned short sR[64 * 34];   // rbf tile bf16
  __shared__ int   sSrc[64];
  __shared__ int   sDst[64];
  __shared__ float sEv[64 * 3];

  const int tid = threadIdx.x, lane = tid & 31, wv = tid >> 5;
  const int e0 = blockIdx.x * 64;

  for (int i = tid; i < 64 * KW; i += 256) {
    int r = i / KW, c = i - r * KW;
    int e = e0 + r;
    float v = (e < nE) ? wgt[(size_t)e * KW + c] : 0.f;
    sA[r * 418 + c] = (unsigned short)f2bf(v);
  }
  for (int i = tid; i < 64 * Rr; i += 256) {
    int r = i >> 5, c = i & 31;
    int e = e0 + r;
    float v = (e < nE) ? erbf[(size_t)e * Rr + c] : 0.f;
    sR[r * 34 + c] = (unsigned short)f2bf(v);
  }
  if (tid < 64) {
    int e = e0 + tid;
    sSrc[tid] = (e < nE) ? srcI[e] : 0;
    sDst[tid] = (e < nE) ? dstI[e] : 0;
  }
  for (int i = tid; i < 64 * 3; i += 256) {
    int r = i / 3, c = i - r * 3;
    int e = e0 + r;
    sEv[i] = (e < nE) ? evec[(size_t)e * 3 + c] : 0.f;
  }
  __syncthreads();

  // GEMM1: h1 = silu(weight @ iw1 + ib1), 64x384, K=416 (13 K-tiles)
  v8f acc1[3][4];
#pragma unroll
  for (int j = 0; j < 3; ++j)
#pragma unroll
    for (int rt = 0; rt < 4; ++rt) acc1[j][rt] = v8f_zero();

  for (int kt = 0; kt < 13; ++kt) {
    v16bf a[4];
#pragma unroll
    for (int rt = 0; rt < 4; ++rt) a[rt] = load_a_lds(sA, rt * 16, 418, kt * 32, lane);
#pragma unroll
    for (int j = 0; j < 3; ++j) {
      int ct = wv + 8 * j;
      if (kt + 1 < 13)
        __builtin_prefetch(iw1s + ((size_t)(ct * 13 + kt + 1) * 32u + (unsigned)lane) * 8u, 0, 1);
      v16bf b = load_b_swz(iw1s, ct, kt, 13, lane);
#pragma unroll
      for (int rt = 0; rt < 4; ++rt) acc1[j][rt] = wmma_bf16(a[rt], b, acc1[j][rt]);
    }
  }
  float b1v[3];
#pragma unroll
  for (int j = 0; j < 3; ++j) b1v[j] = ib1[((wv + 8 * j) << 4) + (lane & 15)];
  __syncthreads();  // everyone done reading weight tile
#pragma unroll
  for (int j = 0; j < 3; ++j) {
    int col = ((wv + 8 * j) << 4) + (lane & 15);
#pragma unroll
    for (int rt = 0; rt < 4; ++rt) {
#pragma unroll
      for (int i = 0; i < 8; ++i) {
        int row = rt * 16 + ((lane >> 4) << 3) + i;
        sA[row * 386 + col] = (unsigned short)f2bf(silu(acc1[j][rt][i] + b1v[j]));
      }
    }
  }
  __syncthreads();

  const float INV3 = 0.57735026918962576f;  // 1/sqrt(3)
  const float INVH = 0.08838834764831845f;  // 1/sqrt(128)

  // GEMM2 (K=384) + rbf GEMM (K=32) + fused epilogue/scatter
#pragma unroll
  for (int j = 0; j < 3; ++j) {
    int ct  = wv + 8 * j;
    int col = (ct << 4) + (lane & 15);     // absolute column in [128*j, 128*j+128)
    float b2v = ib2[col];
    float rbv = rb[col];
    int c = col & 127;
#pragma unroll
    for (int rt = 0; rt < 4; ++rt) {
      v8f acc = v8f_zero();
#pragma unroll
      for (int kt = 0; kt < 12; ++kt) {
        v16bf a = load_a_lds(sA, rt * 16, 386, kt * 32, lane);
        v16bf b = load_b_swz(iw2s, ct, kt, 12, lane);
        acc = wmma_bf16(a, b, acc);
      }
      v8f accR = v8f_zero();
      {
        v16bf ar = load_a_lds(sR, rt * 16, 34, 0, lane);
        v16bf br = load_b_swz(rws, ct, 0, 1, lane);
        accR = wmma_bf16(ar, br, accR);
      }
#pragma unroll
      for (int i = 0; i < 8; ++i) {
        int el = rt * 16 + ((lane >> 4) << 3) + i;
        int e  = e0 + el;
        if (e < nE) {
          float wval = acc[i] + b2v;
          float rval = accR[i] + rbv;
          float rbfh = rval * wval;
          int s  = sSrc[el];
          int dn = sDst[el];
          float mval = xh[(size_t)s * H3 + col] * rbfh;
          if (j == 0) {
            atomAddF(&dx[(size_t)dn * Hh + c], mval);
          } else if (j == 1) {
            float xh2 = mval * INV3 * INVH;
            const float* vp = vec + (size_t)s * H3 + c;
            float* dv = dvec + (size_t)dn * H3 + c;
            atomAddF(dv,          vp[0]       * xh2);
            atomAddF(dv + Hh,     vp[Hh]      * xh2);
            atomAddF(dv + 2 * Hh, vp[2 * Hh]  * xh2);
          } else {
            float xh3 = mval * INVH;
            float* dv = dvec + (size_t)dn * H3 + c;
            atomAddF(dv,          sEv[el * 3 + 0] * xh3);
            atomAddF(dv + Hh,     sEv[el * 3 + 1] * xh3);
            atomAddF(dv + 2 * Hh, sEv[el * 3 + 2] * xh3);
          }
        }
      }
    }
  }
}

// ---------- host ----------

extern "C" void kernel_launch(void* const* d_in, const int* in_sizes, int n_in,
                              void* d_out, int out_size, void* d_ws, size_t ws_size,
                              hipStream_t stream) {
  (void)n_in; (void)out_size; (void)ws_size;
  const float* x    = (const float*)d_in[0];
  const float* vec  = (const float*)d_in[1];
  const int*   eidx = (const int*)d_in[2];
  const float* erbf = (const float*)d_in[3];
  const float* wgt  = (const float*)d_in[4];
  const float* evec = (const float*)d_in[5];
  const float* xw1  = (const float*)d_in[6];
  const float* xb1  = (const float*)d_in[7];
  const float* xw2  = (const float*)d_in[8];
  const float* xb2  = (const float*)d_in[9];
  const float* rw   = (const float*)d_in[10];
  const float* rb   = (const float*)d_in[11];
  const float* iw1  = (const float*)d_in[12];
  const float* ib1  = (const float*)d_in[13];
  const float* iw2  = (const float*)d_in[14];
  const float* ib2  = (const float*)d_in[15];

  const int nNodes = in_sizes[0] / Hh;
  const int nE     = in_sizes[2] / 2;

  // workspace layout (32B-aligned offsets)
  char* ws = (char*)d_ws;
  unsigned int* iw1s = (unsigned int*)(ws + 0);        // 416*384*2 = 319488
  unsigned int* iw2s = (unsigned int*)(ws + 319488);   // 384*384*2 = 294912
  unsigned int* rws  = (unsigned int*)(ws + 614400);   //  32*384*2 =  24576
  unsigned int* xw1s = (unsigned int*)(ws + 638976);   // 128*128*2 =  32768
  unsigned int* xw2s = (unsigned int*)(ws + 671744);   // 128*384*2 =  98304
  float*        xh   = (float*)(ws + 770048);          // N*384*4

  float* dx   = (float*)d_out;
  float* dvec = dx + (size_t)nNodes * Hh;

  size_t outN = (size_t)nNodes * (Hh + H3);
  zero_kernel<<<2048, 256, 0, stream>>>((float*)d_out, outN);

  swz_b_kernel<<<312, 256, 0, stream>>>(iw1, iw1s, KW, H3);
  swz_b_kernel<<<288, 256, 0, stream>>>(iw2, iw2s, H3, H3);
  swz_b_kernel<<<24,  256, 0, stream>>>(rw,  rws,  Rr, H3);
  swz_b_kernel<<<32,  256, 0, stream>>>(xw1, xw1s, Hh, Hh);
  swz_b_kernel<<<96,  256, 0, stream>>>(xw2, xw2s, Hh, H3);

  node_proj_kernel<<<(nNodes + 63) / 64, 256, 0, stream>>>(x, xw1s, xb1, xw2s, xb2, xh, nNodes);

  const int* srcI = eidx;
  const int* dstI = eidx + nE;
  edge_msg_kernel<<<(nE + 63) / 64, 256, 0, stream>>>(
      wgt, erbf, srcI, dstI, evec, iw1s, ib1, iw2s, ib2, rws, rb, xh, vec, dx, dvec, nE);
}
